// PredictionDecoder_30717606101551
// MI455X (gfx1250) — matrix operand order repaired and checked by reading, not compiled, not run on hardware
//
#include <hip/hip_runtime.h>
#include <hip/hip_bf16.h>

// ---------------- problem constants (match reference) ----------------
#define A_TOTAL     8400
#define C_STRIDE    144          // 64 DFL + 80 class scores
#define NUM_CLASSES 80
#define REG_MAX     16
#define MAX_DET     100
#define PRE_NMS_K   4096
#define CONF_T      0.2f
#define IOU_T       0.7f
#define NEG         (-1e9f)
#define NBUCKET     1024
#define BUCKET_SCALE 160.0f      // covers scores in (0.2, ~6.6)

typedef __attribute__((ext_vector_type(16))) _Float16 v16h;
typedef __attribute__((ext_vector_type(8)))  float    v8f;

// CDNA5 async global->LDS copies (ASYNCcnt-tracked), guarded so either
// toolchain still compiles; fallback is plain VMEM loads.
// Builtin prototypes (from hipcc diagnostics): typed pointers,
//   b32 : (addrspace(1) int*, addrspace(3) int*, imm int, imm int)
//   b128: (addrspace(1) int __vector(4)*, addrspace(3) int __vector(4)*, imm, imm)
#if defined(__HIP_DEVICE_COMPILE__) && defined(__gfx1250__) && \
    __has_builtin(__builtin_amdgcn_global_load_async_to_lds_b128) && \
    __has_builtin(__builtin_amdgcn_global_load_async_to_lds_b32) && \
    __has_builtin(__builtin_amdgcn_s_wait_asynccnt)
#define USE_ASYNC_LDS 1
typedef int v4i_t __attribute__((vector_size(16)));
typedef __attribute__((address_space(1))) int   g_i32;
typedef __attribute__((address_space(3))) int   l_i32;
typedef __attribute__((address_space(1))) v4i_t g_v4i;
typedef __attribute__((address_space(3))) v4i_t l_v4i;
#else
#define USE_ASYNC_LDS 0
#endif

// ---------------- init: zero hist/counters, NEG-fill candidate scores --------
__global__ __launch_bounds__(256) void init_kernel(unsigned* __restrict__ hist,
                                                   unsigned* __restrict__ cnt,
                                                   float* __restrict__ cand_score,
                                                   int B) {
  int i = blockIdx.x * blockDim.x + threadIdx.x;
  if (i < B * NBUCKET)   hist[i] = 0u;
  if (i < B)             cnt[i]  = 0u;
  if (i < B * PRE_NMS_K) cand_score[i] = NEG;
}

// ---------------- decode: DFL expectation via WMMA, boxes in pixel xyxy ------
// One wave handles 16 anchors. B-matrix (32x16 f16): lane n (0..15) = anchor
// column n with K=0..15 (first side of pair); lanes 16..31 = same anchors with
// K=16..31 (second side). A-matrix (16x32 f16) is a constant weight matrix:
//   row 0: 1 for K<16        -> denominator of side 2p
//   row 1: K for K<16        -> numerator   of side 2p
//   row 8: 1 for K>=16       -> denominator of side 2p+1
//   row 9: K-16 for K>=16    -> numerator   of side 2p+1
// C layout: VGPR0 lane l<16 = C[0][l] (den side 2p, anchor l);
//           lane l>=16 = C[8][l-16] (den side 2p+1). VGPR1 = numerators.
__global__ __launch_bounds__(256) void decode_kernel(const float* __restrict__ preds,
                                                     float* __restrict__ boxes) {
  const int b    = blockIdx.x;
  const int lane = threadIdx.x & 31;
  const int wib  = threadIdx.x >> 5;
  const int grp  = blockIdx.y * 8 + wib;          // 16-anchor group id
  const int baseA = grp * 16;
  if (baseA >= A_TOTAL) return;                   // wave-uniform exit
  const int n     = lane & 15;                    // anchor column / M row
  const int khalf = lane >> 4;                    // which K half this lane holds
  const int a     = baseA + n;
  const float* p  = preds + ((size_t)b * A_TOTAL + a) * C_STRIDE;

  // constant A (weights), contiguous K order (K = 16*khalf + i), matching B
  v16h aW;
#pragma unroll
  for (int i = 0; i < 16; ++i) {
    int K = khalf * 16 + i;
    float w = 0.0f;
    if      (n == 0) w = (K < 16)  ? 1.0f            : 0.0f;
    else if (n == 1) w = (K < 16)  ? (float)K        : 0.0f;
    else if (n == 8) w = (K >= 16) ? 1.0f            : 0.0f;
    else if (n == 9) w = (K >= 16) ? (float)(K - 16) : 0.0f;
    aW[i] = (_Float16)w;
  }

  float dist01, dist23;
#pragma unroll
  for (int pass = 0; pass < 2; ++pass) {
    const int side = 2 * pass + khalf;            // 0:l 1:t 2:r 3:b
    // 16 contiguous f32, 16B-aligned (row stride 576B, side stride 64B)
    const float4* q4 = (const float4*)(p + side * REG_MAX);
    float x[16];
#pragma unroll
    for (int v = 0; v < 4; ++v) {
      float4 f = q4[v];
      x[4 * v + 0] = f.x; x[4 * v + 1] = f.y;
      x[4 * v + 2] = f.z; x[4 * v + 3] = f.w;
    }
    float m = -1e30f;
#pragma unroll
    for (int i = 0; i < 16; ++i) m = fmaxf(m, x[i]);
    v16h bE;
#pragma unroll
    for (int i = 0; i < 16; ++i) bE[i] = (_Float16)__expf(x[i] - m);

    v8f c = {};
    c = __builtin_amdgcn_wmma_f32_16x16x32_f16(false, aW, false, bE,
                                               (short)0, c, false, false);
    float d = c[1] / c[0];                        // DFL expectation num/den
    if (pass == 0) dist01 = d; else dist23 = d;
  }

  // anchor center + stride; constant divisors per pyramid level (no v_rcp path)
  int row, col; float s;
  if (a < 6400)      { int ai = a;        s = 8.0f;  row = ai / 80; col = ai % 80; }
  else if (a < 8000) { int ai = a - 6400; s = 16.0f; row = ai / 40; col = ai % 40; }
  else               { int ai = a - 8000; s = 32.0f; row = ai / 20; col = ai % 20; }
  const float ac = (khalf == 0) ? (col + 0.5f) : (row + 0.5f);

  // lanes 0..15: (x0,x1) from (l,r);  lanes 16..31: (y0,y1) from (t,b)
  const float c0 = (ac - dist01) * s;
  const float c1 = (ac + dist23) * s;
  const float o0 = __shfl_xor(c0, 16, 32);
  const float o1 = __shfl_xor(c1, 16, 32);
  if (lane < 16) {
    float4* dst = (float4*)(boxes + ((size_t)b * A_TOTAL + a) * 4);
    *dst = make_float4(c0, o0, c1, o1);           // x0,y0,x1,y1
  }
}

// ---------------- histogram of class scores > CONF_T -------------------------
__device__ __forceinline__ int score_bucket(float s) {
  int bk = (int)((s - CONF_T) * BUCKET_SCALE);
  return bk < 0 ? 0 : (bk > NBUCKET - 1 ? NBUCKET - 1 : bk);
}

__global__ __launch_bounds__(256) void hist_kernel(const float* __restrict__ preds,
                                                   unsigned* __restrict__ hist) {
  __shared__ unsigned lh[NBUCKET];
  for (int i = threadIdx.x; i < NBUCKET; i += 256) lh[i] = 0u;
  __syncthreads();
  const int b = blockIdx.y;
  const size_t per  = (size_t)A_TOTAL * NUM_CLASSES;
  const size_t span = (size_t)gridDim.x * 256;
  const float* base = preds + (size_t)b * A_TOTAL * C_STRIDE;
#pragma unroll
  for (int k = 0; k < 4; ++k) {
    size_t idx = (size_t)blockIdx.x * 256 + threadIdx.x + (size_t)k * span;
    if (idx < per) {
      size_t a = idx / NUM_CLASSES, c = idx % NUM_CLASSES;
      const float* addr = base + a * C_STRIDE + 64 + c;
      __builtin_prefetch(addr + 8 * C_STRIDE, 0, 1);   // global_prefetch_b8
      float s = *addr;
      if (s > CONF_T) atomicAdd(&lh[score_bucket(s)], 1u);
    }
  }
  __syncthreads();
  for (int i = threadIdx.x; i < NBUCKET; i += 256)
    if (lh[i]) atomicAdd(&hist[(size_t)b * NBUCKET + i], lh[i]);
}

// ---------------- suffix-scan: per-image bucket threshold for top-K ----------
__global__ void scan_kernel(const unsigned* __restrict__ hist,
                            int* __restrict__ tbuck) {
  const int b = blockIdx.x;
  if (threadIdx.x == 0) {
    unsigned acc = 0; int t = 0;
    for (int i = NBUCKET - 1; i >= 0; --i) {
      acc += hist[(size_t)b * NBUCKET + i];
      if (acc >= PRE_NMS_K) { t = i; break; }
    }
    tbuck[b] = t;
  }
}

// ---------------- compaction: gather top-K candidates ------------------------
__global__ __launch_bounds__(256) void compact_kernel(
    const float* __restrict__ preds, const float* __restrict__ boxes,
    const int* __restrict__ tbuck, unsigned* __restrict__ cnt,
    float* __restrict__ cand_score, float* __restrict__ cand_box,
    int* __restrict__ cand_cls) {
  const int b  = blockIdx.y;
  const int tb = tbuck[b];
  const size_t per  = (size_t)A_TOTAL * NUM_CLASSES;
  const size_t span = (size_t)gridDim.x * 256;
  const float* base = preds + (size_t)b * A_TOTAL * C_STRIDE;
#pragma unroll
  for (int k = 0; k < 4; ++k) {
    size_t idx = (size_t)blockIdx.x * 256 + threadIdx.x + (size_t)k * span;
    if (idx < per) {
      size_t a = idx / NUM_CLASSES, c = idx % NUM_CLASSES;
      const float* addr = base + a * C_STRIDE + 64 + c;
      __builtin_prefetch(addr + 8 * C_STRIDE, 0, 1);
      float s = *addr;
      if (s > CONF_T && score_bucket(s) >= tb) {
        unsigned pos = atomicAdd(&cnt[b], 1u);
        if (pos < PRE_NMS_K) {
          size_t g = (size_t)b * PRE_NMS_K + pos;
          cand_score[g] = s;
          cand_cls[g]   = (int)c;
          const float4 bx = *(const float4*)(boxes + ((size_t)b * A_TOTAL + a) * 4);
          *(float4*)(cand_box + g * 4) = bx;
        }
      }
    }
  }
}

// ---------------- greedy class-aware NMS: one block per image, all in LDS ----
#define NMS_THREADS 1024
#define NMS_ITEMS   (PRE_NMS_K / NMS_THREADS)

__global__ __launch_bounds__(NMS_THREADS) void nms_kernel(
    const float* __restrict__ cand_score, const float* __restrict__ cand_box,
    const int* __restrict__ cand_cls,
    float* __restrict__ out_boxes, float* __restrict__ out_scores,
    int* __restrict__ out_cls, int* __restrict__ out_num) {
  const int b = blockIdx.x;
  __shared__ float  sc[PRE_NMS_K];     // 16 KB
  __shared__ float4 bxy[PRE_NMS_K];    // 64 KB
  __shared__ int    cl[PRE_NMS_K];     // 16 KB
  __shared__ float  rv[32];
  __shared__ int    ri[32];
  __shared__ float  selBox[4];
  __shared__ int    selInfo[2];        // [0]=index [1]=class

  const int tid  = threadIdx.x;
  const int lane = tid & 31, wid = tid >> 5;

  // stage candidates into LDS; on CDNA5 use async global->LDS DMA
#pragma unroll
  for (int k = 0; k < NMS_ITEMS; ++k) {
    int i = tid + k * NMS_THREADS;
    size_t g = (size_t)b * PRE_NMS_K + i;
#if USE_ASYNC_LDS
    __builtin_amdgcn_global_load_async_to_lds_b32(
        (g_i32*)(cand_score + g), (l_i32*)&sc[i], 0, 0);
    __builtin_amdgcn_global_load_async_to_lds_b128(
        (g_v4i*)(cand_box + g * 4), (l_v4i*)&bxy[i], 0, 0);
    __builtin_amdgcn_global_load_async_to_lds_b32(
        (g_i32*)(cand_cls + g), (l_i32*)&cl[i], 0, 0);
#else
    sc[i]  = cand_score[g];
    bxy[i] = *(const float4*)(cand_box + g * 4);
    cl[i]  = cand_cls[g];
#endif
  }
#if USE_ASYNC_LDS
  __builtin_amdgcn_s_wait_asynccnt(0);
#endif
  __syncthreads();

  int ndet = 0;
  for (int it = 0; it < MAX_DET; ++it) {
    // block-wide argmax over live scores
    float bv = -1e30f; int bi = 0;
#pragma unroll
    for (int k = 0; k < NMS_ITEMS; ++k) {
      int i = tid + k * NMS_THREADS;
      float v = sc[i];
      if (v > bv) { bv = v; bi = i; }
    }
#pragma unroll
    for (int off = 16; off > 0; off >>= 1) {
      float ov = __shfl_down(bv, off, 32);
      int   oi = __shfl_down(bi, off, 32);
      if (ov > bv) { bv = ov; bi = oi; }
    }
    if (lane == 0) { rv[wid] = bv; ri[wid] = bi; }
    __syncthreads();
    if (wid == 0) {
      bv = rv[lane]; bi = ri[lane];
#pragma unroll
      for (int off = 16; off > 0; off >>= 1) {
        float ov = __shfl_down(bv, off, 32);
        int   oi = __shfl_down(bi, off, 32);
        if (ov > bv) { bv = ov; bi = oi; }
      }
      if (lane == 0) {
        rv[0] = bv; ri[0] = bi;
        if (bv > -5e8f) {
          float4 sb = bxy[bi];
          selBox[0] = sb.x; selBox[1] = sb.y;
          selBox[2] = sb.z; selBox[3] = sb.w;
          selInfo[0] = bi;  selInfo[1] = cl[bi];
        }
      }
    }
    __syncthreads();

    const float bestV = rv[0];
    if (bestV > -5e8f) {
      ++ndet;
      const float sx0 = selBox[0], sy0 = selBox[1];
      const float sx1 = selBox[2], sy1 = selBox[3];
      const int   scl = selInfo[1];
      if (tid == 0) {
        size_t o = (size_t)b * MAX_DET + it;
        out_boxes[o * 4 + 0] = sx0; out_boxes[o * 4 + 1] = sy0;
        out_boxes[o * 4 + 2] = sx1; out_boxes[o * 4 + 3] = sy1;
        out_scores[o] = bestV;
        out_cls[o]    = scl;
        sc[selInfo[0]] = NEG;     // always kill the selected entry
      }
      const float sarea = (sx1 - sx0) * (sy1 - sy0);
#pragma unroll
      for (int k = 0; k < NMS_ITEMS; ++k) {
        int i = tid + k * NMS_THREADS;
        if (cl[i] == scl && sc[i] > -5e8f) {
          float4 q = bxy[i];
          float ix0 = fmaxf(sx0, q.x), iy0 = fmaxf(sy0, q.y);
          float ix1 = fminf(sx1, q.z), iy1 = fminf(sy1, q.w);
          float inter = fmaxf(ix1 - ix0, 0.0f) * fmaxf(iy1 - iy0, 0.0f);
          float area2 = (q.z - q.x) * (q.w - q.y);
          float iou   = inter / (sarea + area2 - inter + 1e-9f);
          if (iou > IOU_T) sc[i] = NEG;
        }
      }
    } else {
      if (tid == 0) {
        size_t o = (size_t)b * MAX_DET + it;
        out_boxes[o * 4 + 0] = -1.0f; out_boxes[o * 4 + 1] = -1.0f;
        out_boxes[o * 4 + 2] = -1.0f; out_boxes[o * 4 + 3] = -1.0f;
        out_scores[o] = -1.0f;
        out_cls[o]    = -1;
      }
    }
    __syncthreads();
  }
  if (tid == 0) out_num[b] = ndet;
}

// ---------------- launch ----------------
extern "C" void kernel_launch(void* const* d_in, const int* in_sizes, int n_in,
                              void* d_out, int out_size, void* d_ws, size_t ws_size,
                              hipStream_t stream) {
  const float* preds = (const float*)d_in[0];
  // d_in[1] (images) only provides shape 640x640; unused on device.
  const int B = in_sizes[0] / (A_TOTAL * C_STRIDE);

  // workspace carve-up
  char* ws = (char*)d_ws;
  float*    boxes      = (float*)ws;    ws += sizeof(float)    * (size_t)B * A_TOTAL * 4;
  unsigned* hist       = (unsigned*)ws; ws += sizeof(unsigned) * (size_t)B * NBUCKET;
  int*      tbuck      = (int*)ws;      ws += sizeof(int)      * (size_t)B;
  unsigned* cnt        = (unsigned*)ws; ws += sizeof(unsigned) * (size_t)B;
  float*    cand_score = (float*)ws;    ws += sizeof(float)    * (size_t)B * PRE_NMS_K;
  int*      cand_cls   = (int*)ws;      ws += sizeof(int)      * (size_t)B * PRE_NMS_K;
  float*    cand_box   = (float*)ws;    ws += sizeof(float)    * (size_t)B * PRE_NMS_K * 4;

  // output carve-up: boxes [B,100,4] f32, scores [B,100] f32,
  // classes [B,100] i32, num_det [B] i32 (flat concat in return order)
  float* out_boxes  = (float*)d_out;
  float* out_scores = out_boxes + (size_t)B * MAX_DET * 4;
  int*   out_cls    = (int*)(out_scores + (size_t)B * MAX_DET);
  int*   out_num    = out_cls + (size_t)B * MAX_DET;

  {
    int n = B * PRE_NMS_K;
    init_kernel<<<(n + 255) / 256, 256, 0, stream>>>(hist, cnt, cand_score, B);
  }
  {
    dim3 g(B, (A_TOTAL / 16 + 7) / 8);   // 525 wave-groups, 8 waves/block
    decode_kernel<<<g, 256, 0, stream>>>(preds, boxes);
  }
  {
    dim3 g(657, B);                      // 657*256*4 >= 8400*80
    hist_kernel<<<g, 256, 0, stream>>>(preds, hist);
  }
  scan_kernel<<<B, 32, 0, stream>>>(hist, tbuck);
  {
    dim3 g(657, B);
    compact_kernel<<<g, 256, 0, stream>>>(preds, boxes, tbuck, cnt,
                                          cand_score, cand_box, cand_cls);
  }
  nms_kernel<<<B, NMS_THREADS, 0, stream>>>(cand_score, cand_box, cand_cls,
                                            out_boxes, out_scores, out_cls, out_num);
}
